// AuxiliaryLossFreeRouter_90744069029990
// MI455X (gfx1250) — matrix operand order repaired and checked by reading, not compiled, not run on hardware
//
#include <hip/hip_runtime.h>
#include <hip/hip_bf16.h>
#include <math.h>

// ---------------- CDNA5 WMMA types ----------------
typedef __attribute__((ext_vector_type(16))) __bf16        v16bf;
typedef __attribute__((ext_vector_type(8)))  float         v8f;
typedef __attribute__((ext_vector_type(8)))  unsigned int  v8u;

#define D_MODEL 2048
#define N_EXP   256
#define TOPK    8
#define TPB     16      // tokens per block (one WMMA M-tile)
#define KCHUNK  64      // K per LDS stage (2 WMMA K-steps)
#define NCHUNK  (D_MODEL / KCHUNK)   // 32

// output layout (floats): rw[131072] | idx[131072] | counts[256] | lb | max | min | expected
#define OFF_IDX    131072
#define OFF_CNT    262144
#define OFF_LB     262400
#define OFF_MAX    262401
#define OFF_MIN    262402
#define OFF_EXP    262403

__device__ __forceinline__ unsigned short f32_to_bf16(float f) {
  unsigned int u = __builtin_bit_cast(unsigned int, f);
  unsigned int r = u + 0x7FFFu + ((u >> 16) & 1u);   // round-to-nearest-even
  return (unsigned short)(r >> 16);
}

__device__ __forceinline__ unsigned int pack_bf16x2(float a, float b) {
  return (unsigned int)f32_to_bf16(a) | ((unsigned int)f32_to_bf16(b) << 16);
}

// ---------------- kernel 0: convert gate_w to bf16 (vectorized), zero the counts ----------------
__global__ void cvt_w_kernel(const float* __restrict__ w,
                             unsigned short* __restrict__ wb,
                             float* __restrict__ out) {
  int i = blockIdx.x * blockDim.x + threadIdx.x;       // one float4 per thread
  if (i < (N_EXP * D_MODEL) / 4) {
    float4 v = reinterpret_cast<const float4*>(w)[i];
    uint2 p;
    p.x = pack_bf16x2(v.x, v.y);
    p.y = pack_bf16x2(v.z, v.w);
    reinterpret_cast<uint2*>(wb)[i] = p;
  }
  if (i < N_EXP) out[OFF_CNT + i] = 0.0f;
}

// ---------------- kernel 1: gate GEMM (WMMA bf16) + top-8 + softmax + histogram ----------------
__global__ __launch_bounds__(256) void router_kernel(
    const float* __restrict__ x,
    const unsigned short* __restrict__ wb,
    const float* __restrict__ bias,
    float* __restrict__ out) {

  // double-buffered x chunk, stored as packed bf16 pairs: 2 * 16 * 32 dwords = 4 KB
  __shared__ __align__(16) unsigned int As[2][TPB][KCHUNK / 2];
  __shared__ float Ls[TPB][N_EXP + 8];                       // ~16.5 KB logits

  const int tid  = threadIdx.x;
  const int lane = tid & 31;
  const int wave = tid >> 5;          // 0..7
  const long t0  = (long)blockIdx.x * TPB;

  const int mRow = lane & 15;
  const int half = lane >> 4;         // 0/1: which K-half this lane holds
  const int nCol = lane & 15;         // expert column within N-tile
  const int e0   = wave * 32;         // each wave: experts [e0, e0+32)

  // staging map: thread i covers row (i>>4), 4 consecutive k at ((i&15)*4)
  const int sm = tid >> 4;
  const int sk = (tid & 15) * 4;
  const float* xrow = x + (t0 + sm) * D_MODEL + sk;

  // bias folded into the accumulators (same value for every C row of a lane)
  const float ba = bias[e0 + nCol];
  const float bb = bias[e0 + 16 + nCol];
  v8f acc0, acc1;
  #pragma unroll
  for (int r = 0; r < 8; ++r) { acc0[r] = ba; acc1[r] = bb; }

  // stage one 16x64 chunk: float4 load -> 2 packed bf16 dwords -> ds_store_b64
  auto stage = [&](int buf, int kbase) {
    float4 v = *reinterpret_cast<const float4*>(xrow + kbase);
    uint2 p;
    p.x = pack_bf16x2(v.x, v.y);
    p.y = pack_bf16x2(v.z, v.w);
    *reinterpret_cast<uint2*>(&As[buf][sm][sk >> 1]) = p;
  };

  stage(0, 0);
  __syncthreads();

  for (int c = 0; c < NCHUNK; ++c) {
    const int cur = c & 1;
    if (c + 1 < NCHUNK) {
      stage(cur ^ 1, (c + 1) * KCHUNK);
      if (c + 2 < NCHUNK)
        __builtin_prefetch(xrow + (c + 2) * KCHUNK, 0, 0);   // global_prefetch_b8
    }

    #pragma unroll
    for (int s = 0; s < 2; ++s) {                // two K=32 WMMA steps per chunk
      // A fragment from LDS: 16 contiguous k (k = 16*half + i), matching B
      v8u a_u;
      {
        uint4 lo = *reinterpret_cast<const uint4*>(&As[cur][mRow][s * 16 + half * 8]);
        uint4 hi = *reinterpret_cast<const uint4*>(&As[cur][mRow][s * 16 + half * 8 + 4]);
        a_u[0] = lo.x; a_u[1] = lo.y; a_u[2] = lo.z; a_u[3] = lo.w;
        a_u[4] = hi.x; a_u[5] = hi.y; a_u[6] = hi.z; a_u[7] = hi.w;
      }
      v16bf a = __builtin_bit_cast(v16bf, a_u);

      // B fragments: lane = expert column; 16 contiguous bf16 (32 B) per lane
      long kg    = (long)c * KCHUNK + s * 32 + half * 16;
      long base0 = (long)(e0 + nCol)      * D_MODEL + kg;
      long base1 = (long)(e0 + 16 + nCol) * D_MODEL + kg;
      v16bf b0 = __builtin_bit_cast(v16bf, *reinterpret_cast<const v8u*>(wb + base0));
      v16bf b1 = __builtin_bit_cast(v16bf, *reinterpret_cast<const v8u*>(wb + base1));

      acc0 = __builtin_amdgcn_wmma_f32_16x16x32_bf16(false, a, false, b0,
                                                     (short)0, acc0, false, false);
      acc1 = __builtin_amdgcn_wmma_f32_16x16x32_bf16(false, a, false, b1,
                                                     (short)0, acc1, false, false);
    }
    __syncthreads();   // single barrier per chunk (ping-pong buffers)
  }

  // spill logits into LDS: C layout VGPR r -> M = r + 8*half, N = nCol
  {
    int ea = e0 + nCol, eb = e0 + 16 + nCol;
    #pragma unroll
    for (int r = 0; r < 8; ++r) {
      int m = r + half * 8;
      Ls[m][ea] = acc0[r];
      Ls[m][eb] = acc1[r];
    }
  }
  __syncthreads();

  // top-8 + softmax: wave w owns tokens {2w, 2w+1}
  #pragma unroll
  for (int tt = 0; tt < 2; ++tt) {
    int m = wave * 2 + tt;
    float v[8];
    #pragma unroll
    for (int j = 0; j < 8; ++j) v[j] = Ls[m][lane + 32 * j];

    float topv[TOPK];
    int   tope[TOPK];
    #pragma unroll
    for (int it = 0; it < TOPK; ++it) {
      float best = -INFINITY; int bj = 0;
      #pragma unroll
      for (int j = 0; j < 8; ++j)
        if (v[j] > best) { best = v[j]; bj = j; }
      int be = lane + 32 * bj;
      // wave32 xor-butterfly argmax (all lanes converge to the same result)
      #pragma unroll
      for (int off = 16; off >= 1; off >>= 1) {
        float ov = __shfl_xor(best, off, 32);
        int   oe = __shfl_xor(be,   off, 32);
        if (ov > best || (ov == best && oe < be)) { best = ov; be = oe; }
      }
      topv[it] = best; tope[it] = be;
      // owning lane retires the winner (static-index select chain)
      int jz = be >> 5;
      bool mine = ((be & 31) == lane);
      #pragma unroll
      for (int j = 0; j < 8; ++j)
        if (mine && j == jz) v[j] = -INFINITY;
    }

    // softmax over the 8 selected logits (topv[0] is the max)
    float mx = topv[0];
    float s = 0.0f, wv[TOPK];
    #pragma unroll
    for (int j = 0; j < TOPK; ++j) { wv[j] = __expf(topv[j] - mx); s += wv[j]; }
    float inv = 1.0f / s;

    if (lane == 0) {
      long t = t0 + m;
      #pragma unroll
      for (int j = 0; j < TOPK; ++j) {
        out[t * TOPK + j]           = wv[j] * inv;
        out[OFF_IDX + t * TOPK + j] = (float)tope[j];
        atomicAdd(&out[OFF_CNT + tope[j]], 1.0f);
      }
    }
  }
}

// ---------------- kernel 2: load-balance statistics ----------------
__global__ void stats_kernel(float* __restrict__ out) {
  __shared__ float s_sum[256], s_sq[256], s_mx[256], s_mn[256];
  int t = threadIdx.x;
  float c = out[OFF_CNT + t];
  s_sum[t] = c; s_sq[t] = c * c; s_mx[t] = c; s_mn[t] = c;
  __syncthreads();
  for (int off = 128; off >= 1; off >>= 1) {
    if (t < off) {
      s_sum[t] += s_sum[t + off];
      s_sq[t]  += s_sq[t + off];
      s_mx[t]   = fmaxf(s_mx[t], s_mx[t + off]);
      s_mn[t]   = fminf(s_mn[t], s_mn[t + off]);
    }
    __syncthreads();
  }
  if (t == 0) {
    float mean = s_sum[0] / 256.0f;
    float var  = (s_sq[0] - 256.0f * mean * mean) / 255.0f;   // ddof=1
    float sd   = sqrtf(fmaxf(var, 0.0f));
    out[OFF_LB]  = sd / (mean + 1e-6f);
    out[OFF_MAX] = s_mx[0];
    out[OFF_MIN] = s_mn[0];
    out[OFF_EXP] = (float)(16384 * TOPK) / (float)N_EXP;      // 512
  }
}

// ---------------- launch ----------------
extern "C" void kernel_launch(void* const* d_in, const int* in_sizes, int n_in,
                              void* d_out, int out_size, void* d_ws, size_t ws_size,
                              hipStream_t stream) {
  const float* x    = (const float*)d_in[0];
  const float* gw   = (const float*)d_in[1];
  const float* bias = (const float*)d_in[2];
  float* out        = (float*)d_out;
  unsigned short* wb = (unsigned short*)d_ws;   // bf16 gate_w: 256*2048*2 = 1 MB

  const int nTok = in_sizes[0] / D_MODEL;       // 16384

  cvt_w_kernel<<<(N_EXP * D_MODEL / 4 + 255) / 256, 256, 0, stream>>>(gw, wb, out);
  router_kernel<<<nTok / TPB, 256, 0, stream>>>(x, wb, bias, out);
  stats_kernel<<<1, 256, 0, stream>>>(out);
}
